// FirstBlockTexture_10763188043862
// MI455X (gfx1250) — compile-verified, matrix operand order checked by reading, not run on hardware
//
#include <hip/hip_runtime.h>

#define NV_   100000
#define NF_   200000
#define T_    600000
#define COUT_ 64
#define CINV_ 128            // 2*COUT
#define CD_   256            // CIN_V * DM
#define KS_   27
#define KPAD_ 32

typedef __attribute__((ext_vector_type(16))) __bf16 v16bf;
typedef __attribute__((ext_vector_type(8)))  float  v8f;

struct U16x16 { uint4 lo, hi; };

__device__ __forceinline__ unsigned short f2bf(float f) {
    unsigned u = __builtin_bit_cast(unsigned, f);
    u += 0x7fffu + ((u >> 16) & 1u);               // round-to-nearest-even
    return (unsigned short)(u >> 16);
}
__device__ __forceinline__ float bf2f(unsigned short h) {
    unsigned u = ((unsigned)h) << 16;
    return __builtin_bit_cast(float, u);
}

// A-fragment (16x32 bf16, row-major source): lane m=lane&15 holds row m.
// h=0 lanes: K {0..7, 16..23}; h=1 lanes: K {8..15, 24..31}.
__device__ __forceinline__ v16bf load_a_frag(const unsigned short* __restrict__ row, int h) {
    U16x16 t;
    t.lo = *(const uint4*)(row + 8 * h);
    t.hi = *(const uint4*)(row + 16 + 8 * h);
    return __builtin_bit_cast(v16bf, t);
}
// B-fragment: pre-swizzled, 16 contiguous bf16 per lane.
__device__ __forceinline__ v16bf load_b_frag(const unsigned short* __restrict__ p) {
    const uint4* q = (const uint4*)p;
    U16x16 t; t.lo = q[0]; t.hi = q[1];
    return __builtin_bit_cast(v16bf, t);
}
__device__ __forceinline__ v8f wmma_bf16(v16bf a, v16bf b, v8f c) {
    return __builtin_amdgcn_wmma_f32_16x16x32_bf16(false, a, false, b, (short)0, c, false, false);
}

// ---------------- Kernel A: fused texel conv + geometry MLP -> net[NF,128] f32
__global__ void net_kernel(const float* __restrict__ geo, const float* __restrict__ tex,
                           const float* __restrict__ bary, const int* __restrict__ num_tex,
                           const float* __restrict__ Wf2f, const float* __restrict__ bF2F,
                           const float* __restrict__ Wmlp, const float* __restrict__ bMLP,
                           float* __restrict__ net) {
    int gid = blockIdx.x * 256 + threadIdx.x;
    if (gid >= NF_ * COUT_) return;
    int f = gid >> 6, c = gid & 63;
    float wt[9];
#pragma unroll
    for (int ij = 0; ij < 9; ++ij) wt[ij] = Wf2f[ij * COUT_ + c];
    float acc = 0.f;
    int tb = f * 3;
#pragma unroll
    for (int t = 0; t < 3; ++t) {
        const float* by = bary + (size_t)(tb + t) * 3;
        const float* tx = tex + (size_t)(tb + t) * 3;
#pragma unroll
        for (int i = 0; i < 3; ++i)
#pragma unroll
            for (int j = 0; j < 3; ++j)
                acc += by[i] * tx[j] * wt[3 * i + j];
    }
    float Tn = fmaxf(acc / (float)num_tex[f] + bF2F[c], 0.f);
    float ag = 0.f;
#pragma unroll
    for (int g = 0; g < 8; ++g) ag += geo[(size_t)f * 8 + g] * Wmlp[g * COUT_ + c];
    float Gn = fmaxf(ag + bMLP[c], 0.f);
    net[(size_t)f * CINV_ + c] = Tn;
    net[(size_t)f * CINV_ + COUT_ + c] = Gn;
}

// ---------------- Kernel A2: pad filt_coeff 27->32, convert to bf16
__global__ void filt_pad_kernel(const float* __restrict__ filt, unsigned short* __restrict__ filt_bf) {
    int gid = blockIdx.x * 256 + threadIdx.x;
    if (gid >= NF_ * KPAD_) return;
    int f = gid >> 5, k = gid & 31;
    float v = (k < KS_) ? filt[(size_t)f * KS_ + k] : 0.f;
    filt_bf[gid] = f2bf(v);
}

// ---------------- Pack W_dw [27,256] -> WMMA B fragments [16 ntiles][32 lanes][16]
__global__ void pack_wdw_kernel(const float* __restrict__ Wdw, unsigned short* __restrict__ Bdw) {
    int tid = blockIdx.x * blockDim.x + threadIdx.x;
    if (tid >= 16 * 32) return;
    int lane = tid & 31, n = lane & 15, h = lane >> 4;
    int nt = tid >> 5, col = nt * 16 + n;
#pragma unroll
    for (int j = 0; j < 16; ++j) {
        int k = (j < 8) ? (8 * h + j) : (16 + 8 * h + (j - 8));
        float v = (k < KS_) ? Wdw[k * CD_ + col] : 0.f;
        Bdw[tid * 16 + j] = f2bf(v);
    }
}

// ---------------- Pack W_pw [256,64] -> [4 ntiles][8 ksteps][32 lanes][16]
__global__ void pack_wpw_kernel(const float* __restrict__ Wpw, unsigned short* __restrict__ Bpw) {
    int tid = blockIdx.x * blockDim.x + threadIdx.x;
    if (tid >= 4 * 8 * 32) return;
    int lane = tid & 31, n = lane & 15, h = lane >> 4;
    int ks = (tid >> 5) & 7, nt = tid >> 8;
    int col = nt * 16 + n;
#pragma unroll
    for (int j = 0; j < 16; ++j) {
        int k = ks * 32 + ((j < 8) ? (8 * h + j) : (16 + 8 * h + (j - 8)));
        Bpw[tid * 16 + j] = f2bf(Wpw[k * COUT_ + col]);
    }
}

// ---------------- Kernel B: M = filt @ Wdw via WMMA, contrib = M * net broadcast
__global__ __launch_bounds__(32) void mix_kernel(const unsigned short* __restrict__ filt_bf,
                                                 const unsigned short* __restrict__ Bdw,
                                                 const float* __restrict__ net,
                                                 unsigned short* __restrict__ contrib_bf) {
    int base = blockIdx.x * 16;
    int lane = threadIdx.x, m = lane & 15, h = lane >> 4;
    v16bf a = load_a_frag(filt_bf + (size_t)(base + m) * KPAD_, h);
#pragma unroll 4
    for (int nt = 0; nt < 16; ++nt) {
        v16bf b = load_b_frag(Bdw + (size_t)(nt * 32 + lane) * 16);
        v8f c = {0.f, 0.f, 0.f, 0.f, 0.f, 0.f, 0.f, 0.f};
        c = wmma_bf16(a, b, c);
        int col = nt * 16 + m;     // C frag: N = lane&15
        int ch = col >> 1;         // contrib col = c*2+d -> net channel = col>>1
#pragma unroll
        for (int r = 0; r < 8; ++r) {
            int mr = r + 8 * h;    // C frag: M = r (+8 for upper lanes)
            float v = c[r] * net[(size_t)(base + mr) * CINV_ + ch];
            contrib_bf[(size_t)(base + mr) * CD_ + col] = f2bf(v);
        }
    }
}

// ---------------- Kernel C: facet->vertex scatter via face table (f32 atomics)
__global__ void scatter_kernel(const unsigned short* __restrict__ contrib_bf,
                               const int* __restrict__ face, float* __restrict__ vfeat) {
    size_t gid = (size_t)blockIdx.x * 256 + threadIdx.x;   // grid = NF blocks exactly
    int f = (int)(gid >> 8), col = (int)(gid & 255);
    float v = bf2f(contrib_bf[gid]);
#pragma unroll
    for (int j = 0; j < 3; ++j) {
        int vtx = face[f * 3 + j];
        atomicAdd(vfeat + (size_t)vtx * CD_ + col, v);
    }
}

// ---------------- Kernel D: mean + convert to bf16
__global__ void vfin_kernel(const float* __restrict__ vfeat, const int* __restrict__ cnt,
                            unsigned short* __restrict__ vfeat_bf) {
    size_t gid = (size_t)blockIdx.x * 256 + threadIdx.x;   // grid = NV blocks
    int v = (int)(gid >> 8);
    int c = cnt[v]; if (c < 1) c = 1;
    vfeat_bf[gid] = f2bf(vfeat[gid] / (float)c);
}

// ---------------- Kernel E: out = relu(vfeat @ Wpw + b) via WMMA (K=256 -> 8 steps)
__global__ __launch_bounds__(32) void pw_kernel(const unsigned short* __restrict__ vfeat_bf,
                                                const unsigned short* __restrict__ Bpw,
                                                const float* __restrict__ bpw,
                                                float* __restrict__ out) {
    int base = blockIdx.x * 16;
    int lane = threadIdx.x, m = lane & 15, h = lane >> 4;
    v8f z = {0.f, 0.f, 0.f, 0.f, 0.f, 0.f, 0.f, 0.f};
    v8f acc[4] = {z, z, z, z};
#pragma unroll
    for (int ks = 0; ks < 8; ++ks) {
        v16bf a = load_a_frag(vfeat_bf + (size_t)(base + m) * CD_ + ks * 32, h);
#pragma unroll
        for (int nt = 0; nt < 4; ++nt) {
            v16bf b = load_b_frag(Bpw + (size_t)((nt * 8 + ks) * 32 + lane) * 16);
            acc[nt] = wmma_bf16(a, b, acc[nt]);
        }
    }
#pragma unroll
    for (int nt = 0; nt < 4; ++nt) {
        int col = nt * 16 + m;
        float bias = bpw[col];
#pragma unroll
        for (int r = 0; r < 8; ++r) {
            int mr = r + 8 * h;
            out[(size_t)(base + mr) * COUT_ + col] = fmaxf(acc[nt][r] + bias, 0.f);
        }
    }
}

extern "C" void kernel_launch(void* const* d_in, const int* in_sizes, int n_in,
                              void* d_out, int out_size, void* d_ws, size_t ws_size,
                              hipStream_t stream) {
    const float* geo      = (const float*)d_in[0];
    const float* tex      = (const float*)d_in[1];
    const float* bary     = (const float*)d_in[2];
    const int*   num_tex  = (const int*)d_in[3];
    const int*   face     = (const int*)d_in[4];
    const int*   nf_count = (const int*)d_in[5];
    // d_in[6] = full_vt_map (unused: face table gives identical incidences)
    const float* filt     = (const float*)d_in[7];
    const float* Wf2f     = (const float*)d_in[8];
    const float* bF2F     = (const float*)d_in[9];
    const float* Wmlp     = (const float*)d_in[10];
    const float* bMLP     = (const float*)d_in[11];
    const float* Wdw      = (const float*)d_in[12];
    const float* Wpw      = (const float*)d_in[13];
    const float* bpw      = (const float*)d_in[14];
    float* out = (float*)d_out;

    char* ws = (char*)d_ws;
    size_t o = 0;
    float*          net        = (float*)(ws + o);          o += (size_t)NF_ * CINV_ * 4;  // 102.4 MB
    unsigned short* filt_bf    = (unsigned short*)(ws + o); o += (size_t)NF_ * KPAD_ * 2;  //  12.8 MB
    unsigned short* contrib_bf = (unsigned short*)(ws + o); o += (size_t)NF_ * CD_ * 2;    // 102.4 MB
    float*          vfeat      = (float*)(ws + o);          o += (size_t)NV_ * CD_ * 4;    // 102.4 MB
    unsigned short* vfeat_bf   = (unsigned short*)(ws + o); o += (size_t)NV_ * CD_ * 2;    //  51.2 MB
    unsigned short* Bdw        = (unsigned short*)(ws + o); o += 16 * 32 * 16 * 2;
    unsigned short* Bpw        = (unsigned short*)(ws + o); o += 4 * 8 * 32 * 16 * 2;

    hipMemsetAsync(vfeat, 0, (size_t)NV_ * CD_ * 4, stream);

    pack_wdw_kernel<<<2, 256, 0, stream>>>(Wdw, Bdw);
    pack_wpw_kernel<<<4, 256, 0, stream>>>(Wpw, Bpw);

    net_kernel<<<(NF_ * COUT_) / 256, 256, 0, stream>>>(geo, tex, bary, num_tex,
                                                        Wf2f, bF2F, Wmlp, bMLP, net);
    filt_pad_kernel<<<(NF_ * KPAD_) / 256, 256, 0, stream>>>(filt, filt_bf);

    mix_kernel<<<NF_ / 16, 32, 0, stream>>>(filt_bf, Bdw, net, contrib_bf);

    scatter_kernel<<<NF_, 256, 0, stream>>>(contrib_bf, face, vfeat);
    vfin_kernel<<<NV_, 256, 0, stream>>>(vfeat, nf_count, vfeat_bf);

    pw_kernel<<<NV_ / 16, 32, 0, stream>>>(vfeat_bf, Bpw, bpw, out);
    (void)in_sizes; (void)n_in; (void)out_size; (void)ws_size; (void)nf_count;
}
